// SelfAttention_72816875536636
// MI455X (gfx1250) — matrix operand order repaired and checked by reading, not compiled, not run on hardware
//
#include <hip/hip_runtime.h>

// ---------------------------------------------------------------------------
// Sigmoid-attention block (no softmax) for MI455X / gfx1250 (wave32, WMMA).
//   qkv  = x @ W_attn + b_attn                    (bf16 WMMA GEMM, f32 accum)
//   S    = sigmoid(QK^T * 0.125) * coulomb        (streamed, no normalization)
//   yatt = S @ V
//   out  = (yatt + x@W_self + b_self) @ W_proj + b_proj
// dist_matrix input is unused by the reference.
// ---------------------------------------------------------------------------

#define N_EMBD 768
#define N_HEAD 12
#define HD     64
#define BZ     4
#define TZ     2048
#define MTOT   (BZ * TZ)   // 8192

typedef __bf16 bf16;
typedef __attribute__((ext_vector_type(16))) __bf16 v16bf;
typedef __attribute__((ext_vector_type(8)))  __bf16 v8bf;
typedef __attribute__((ext_vector_type(4)))  __bf16 v4bf;
typedef __attribute__((ext_vector_type(8)))  float  v8f;

// exact pointee type of the async-to-LDS builtin (from hipcc diagnostic):
// __attribute__((__vector_size__(4 * sizeof(int)))) int
typedef __attribute__((__vector_size__(4 * sizeof(int)))) int v4i32;
typedef __attribute__((address_space(1))) v4i32* gptr_v4i;
typedef __attribute__((address_space(3))) v4i32* lptr_v4i;

#if __has_builtin(__builtin_amdgcn_global_load_async_to_lds_b128)
#define HAVE_ASYNC_LDS 1
#else
#define HAVE_ASYNC_LDS 0
#endif

__device__ __forceinline__ v16bf cat16(v8bf lo, v8bf hi) {
  v16bf r;
#pragma unroll
  for (int j = 0; j < 8; ++j) { r[j] = lo[j]; r[8 + j] = hi[j]; }
  return r;
}

__device__ __forceinline__ v8f wmma_bf16(v16bf a, v16bf b, v8f c) {
  return __builtin_amdgcn_wmma_f32_16x16x32_bf16(false, a, false, b, (short)0, c,
                                                 false, false);
}

// fast sigmoid: v_exp_f32 + v_rcp_f32 (avoid IEEE div expansion)
__device__ __forceinline__ float fast_sigmoid(float x) {
  return __builtin_amdgcn_rcpf(1.0f + __expf(-x));
}

// ---------------------------------------------------------------------------
// Generic bf16-WMMA GEMM: out[M,N] = A[M,K] @ W[K,N] + bias
//   AMODE 0: A = A0 (fp32)        AMODE 1: A = A0 + A1 (fused residual add)
//   OMODE 0: fp32 row-major out   OMODE 1: scatter to q/k/v bf16 [B*nh,T,hd]
// Block: 256 threads (8 waves). Block tile 64(M) x 128(N), K step 32.
// ---------------------------------------------------------------------------
template <int AMODE, int OMODE>
__global__ __launch_bounds__(256) void gemm_bf16_wmma(
    const float* __restrict__ A0, const float* __restrict__ A1,
    const float* __restrict__ W, const float* __restrict__ bias,
    float* __restrict__ outF, bf16* __restrict__ outQ, bf16* __restrict__ outK,
    bf16* __restrict__ outV, int M, int N, int K) {
  const int tid  = threadIdx.x;
  const int lane = tid & 31;
  const int wid  = tid >> 5;
  const int half = lane >> 4;
  const int l16  = lane & 15;
  const int m0   = blockIdx.x * 64;
  const int n0   = blockIdx.y * 128;
  const int waveM = wid & 3;   // 0..3 -> 16-row slice
  const int waveN = wid >> 2;  // 0..1 -> 64-col slice

  __shared__ bf16 lds_a[64][40];    // A tile 64x32 (padded rows, 16B aligned)
  __shared__ bf16 lds_bt[128][40];  // W tile transposed: [n][k]

  v8f acc[4];
#pragma unroll
  for (int s = 0; s < 4; ++s)
#pragma unroll
    for (int i = 0; i < 8; ++i) acc[s][i] = 0.0f;

  for (int k0 = 0; k0 < K; k0 += 32) {
    __syncthreads();
    // ---- stage A tile (64x32 fp32 -> bf16), 2 float4 per thread ----
#pragma unroll
    for (int e = 0; e < 2; ++e) {
      const int f4  = tid * 2 + e;       // 0..511
      const int row = f4 >> 3;           // 8 float4 per row
      const int col = (f4 & 7) * 4;
      float4 v = *(const float4*)&A0[(size_t)(m0 + row) * K + k0 + col];
      if (AMODE == 1) {
        const float4 w = *(const float4*)&A1[(size_t)(m0 + row) * K + k0 + col];
        v.x += w.x; v.y += w.y; v.z += w.z; v.w += w.w;
      }
      v4bf bv;
      bv[0] = (bf16)v.x; bv[1] = (bf16)v.y; bv[2] = (bf16)v.z; bv[3] = (bf16)v.w;
      *(v4bf*)&lds_a[row][col] = bv;
    }
    // ---- stage W tile transposed (32x128 -> [n][k]), 4 float4 per thread ----
#pragma unroll
    for (int e = 0; e < 4; ++e) {
      const int f4  = tid * 4 + e;       // 0..1023
      const int row = f4 >> 5;           // k row, 32 float4 per row
      const int col = (f4 & 31) * 4;     // n col
      const float4 v = *(const float4*)&W[(size_t)(k0 + row) * N + n0 + col];
      lds_bt[col + 0][row] = (bf16)v.x;
      lds_bt[col + 1][row] = (bf16)v.y;
      lds_bt[col + 2][row] = (bf16)v.z;
      lds_bt[col + 3][row] = (bf16)v.w;
    }
    __syncthreads();

    // ---- A fragment: row = waveM*16 + l16; elems j -> K = (16*seg)+half*8+j
    const int r = waveM * 16 + l16;
    const v16bf a = cat16(*(const v8bf*)&lds_a[r][half * 8],
                          *(const v8bf*)&lds_a[r][16 + half * 8]);
#pragma unroll
    for (int s = 0; s < 4; ++s) {
      const int nc = waveN * 64 + s * 16 + l16;
      const v16bf b = cat16(*(const v8bf*)&lds_bt[nc][half * 8],
                            *(const v8bf*)&lds_bt[nc][16 + half * 8]);
      acc[s] = wmma_bf16(a, b, acc[s]);
    }
  }

  // ---- epilogue: C/D layout elem i -> M = i + 8*half, N = l16 ----
#pragma unroll
  for (int s = 0; s < 4; ++s) {
    const int n  = n0 + waveN * 64 + s * 16 + l16;
    const float bv = bias[n];
#pragma unroll
    for (int i = 0; i < 8; ++i) {
      const int m = m0 + waveM * 16 + i + half * 8;
      const float rres = acc[s][i] + bv;
      if (OMODE == 0) {
        outF[(size_t)m * N + n] = rres;
      } else {
        const int which = n / N_EMBD;      // 0=q 1=k 2=v
        const int rem   = n % N_EMBD;
        const int h     = rem >> 6;
        const int d     = rem & 63;
        const int b     = m >> 11;         // / TZ
        const int t     = m & (TZ - 1);
        bf16* dst = (which == 0) ? outQ : ((which == 1) ? outK : outV);
        dst[(((size_t)(b * N_HEAD + h)) * TZ + t) * HD + d] = (bf16)rres;
      }
    }
  }
}

// ---------------------------------------------------------------------------
// Sigmoid attention: Y[b,q,h*64+d] = sum_k sigmoid(q.k/8)*coulomb[b,q,k]*V[k,d]
// Block = one (b,h) x 128 q-rows; 8 waves, each owns a 16-row q-tile.
// Streams k in 32-wide chunks; K tile async-to-LDS + V tile transposed in LDS.
// ---------------------------------------------------------------------------
__global__ __launch_bounds__(256) void attn_sigmoid_wmma(
    const bf16* __restrict__ Q, const bf16* __restrict__ Km,
    const bf16* __restrict__ V, const float* __restrict__ coul,
    float* __restrict__ Y) {
  const int tid  = threadIdx.x;
  const int lane = tid & 31;
  const int wid  = tid >> 5;
  const int half = lane >> 4;
  const int l16  = lane & 15;
  const int bh   = blockIdx.x;          // 0..47
  const int b    = bh / N_HEAD;
  const int h    = bh % N_HEAD;
  const int qr   = blockIdx.y * 128 + wid * 16;

  __shared__ bf16 lds_k[32][72];        // K chunk [k][d]
  __shared__ bf16 lds_vt[64][40];       // V chunk transposed [d][k]
  __shared__ bf16 lds_s[8][16][40];     // per-wave S tile for relayout

  // ---- Q fragments (held in registers across the whole k loop) ----
  const size_t qbase = ((size_t)bh * TZ + (qr + l16)) * HD;
  v16bf aq[2];
#pragma unroll
  for (int dh = 0; dh < 2; ++dh)
    aq[dh] = cat16(*(const v8bf*)&Q[qbase + dh * 32 + half * 8],
                   *(const v8bf*)&Q[qbase + dh * 32 + 16 + half * 8]);

  v8f y[4];
#pragma unroll
  for (int s = 0; s < 4; ++s)
#pragma unroll
    for (int i = 0; i < 8; ++i) y[s][i] = 0.0f;

  const float scale = 0.125f;  // 1/sqrt(64)

  for (int kt = 0; kt < TZ / 32; ++kt) {
    const int k0 = kt * 32;
    __syncthreads();
    {  // cooperative stage: one 16B K row-chunk + one 16B V row-chunk / thread
      const int row  = tid >> 3;        // 0..31 (k index)
      const int dcol = (tid & 7) * 8;   // 0..56 (d index)
      const size_t g = ((size_t)bh * TZ + k0 + row) * HD + dcol;
#if HAVE_ASYNC_LDS
      // CDNA5 async copy: global -> LDS without VGPR round-trip (ASYNCcnt).
      // Builtin signature (from hipcc): (v4i32 AS1*, v4i32 AS3*, imm, imm)
      __builtin_amdgcn_global_load_async_to_lds_b128(
          (gptr_v4i)&Km[g], (lptr_v4i)&lds_k[row][dcol], 0, 0);
#else
      *(v8bf*)&lds_k[row][dcol] = *(const v8bf*)&Km[g];
#endif
      const v8bf vv = *(const v8bf*)&V[g];
#pragma unroll
      for (int e = 0; e < 8; ++e) lds_vt[dcol + e][row] = vv[e];  // transpose
    }
#if HAVE_ASYNC_LDS
    // barrier does not imply async completion; drain our ASYNCcnt first
    asm volatile("s_wait_asynccnt 0x0" ::: "memory");
#endif
    __syncthreads();

    // ---- S = Q @ K^T over the 32-wide k chunk (16q x 32k) ----
    v8f sc[2];
#pragma unroll
    for (int ng = 0; ng < 2; ++ng)
#pragma unroll
      for (int i = 0; i < 8; ++i) sc[ng][i] = 0.0f;
#pragma unroll
    for (int ng = 0; ng < 2; ++ng) {
#pragma unroll
      for (int dh = 0; dh < 2; ++dh) {
        const bf16* kr = &lds_k[ng * 16 + l16][dh * 32 + half * 8];
        const v16bf bk = cat16(*(const v8bf*)kr, *(const v8bf*)(kr + 16));
        sc[ng] = wmma_bf16(aq[dh], bk, sc[ng]);
      }
    }

    // ---- sigmoid * coulomb, relayout 16x32 S tile through per-wave LDS ----
#pragma unroll
    for (int ng = 0; ng < 2; ++ng) {
#pragma unroll
      for (int i = 0; i < 8; ++i) {
        const int qm = qr + i + half * 8;
        const int kn = k0 + ng * 16 + l16;
        const float sv  = sc[ng][i] * scale;
        const float sig = fast_sigmoid(sv);
        // coulomb is 67MB, streamed once per head (12x): non-temporal keeps L2 for
        // the resident q/k/v + activations working set
        const float c =
            __builtin_nontemporal_load(&coul[((size_t)b * TZ + qm) * TZ + kn]);
        lds_s[wid][i + half * 8][ng * 16 + l16] = (bf16)(sig * c);
      }
    }
    // intra-wave LDS RAW: wait for our ds_stores before the relayout reads
    asm volatile("s_wait_dscnt 0x0" ::: "memory");
    const v16bf as = cat16(*(const v8bf*)&lds_s[wid][l16][half * 8],
                           *(const v8bf*)&lds_s[wid][l16][16 + half * 8]);

    // ---- y += S @ V ----
#pragma unroll
    for (int s = 0; s < 4; ++s) {
      const bf16* vr = &lds_vt[s * 16 + l16][half * 8];
      const v16bf bv = cat16(*(const v8bf*)vr, *(const v8bf*)(vr + 16));
      y[s] = wmma_bf16(as, bv, y[s]);
    }
  }

  // ---- write yatt as fp32 [B,T,C] ----
#pragma unroll
  for (int s = 0; s < 4; ++s) {
#pragma unroll
    for (int i = 0; i < 8; ++i) {
      const int qm = qr + i + half * 8;
      const int n  = h * HD + s * 16 + l16;
      Y[((size_t)b * TZ + qm) * N_EMBD + n] = y[s][i];
    }
  }
}

// ---------------------------------------------------------------------------
extern "C" void kernel_launch(void* const* d_in, const int* in_sizes, int n_in,
                              void* d_out, int out_size, void* d_ws,
                              size_t ws_size, hipStream_t stream) {
  const float* x      = (const float*)d_in[0];
  // d_in[1] = dist_matrix (unused by reference)
  const float* coul   = (const float*)d_in[2];
  const float* W_attn = (const float*)d_in[3];
  const float* b_attn = (const float*)d_in[4];
  const float* W_proj = (const float*)d_in[5];
  const float* b_proj = (const float*)d_in[6];
  const float* W_self = (const float*)d_in[7];
  const float* b_self = (const float*)d_in[8];
  float* out = (float*)d_out;

  char* ws = (char*)d_ws;
  // q/k/v bf16 head-major [B*nh, T, hd]: 6,291,456 elems each
  bf16*  q_ws   = (bf16*)(ws);
  bf16*  k_ws   = (bf16*)(ws + 12582912);
  bf16*  v_ws   = (bf16*)(ws + 25165824);
  float* self_y = (float*)(ws + 37748736);  // fp32 [8192,768]
  float* att_y  = (float*)(ws + 62914560);  // fp32 [8192,768]

  const dim3 blk(256);

  // 1) qkv = x @ W_attn + b_attn  -> scatter bf16 q/k/v
  gemm_bf16_wmma<0, 1><<<dim3(MTOT / 64, (3 * N_EMBD) / 128), blk, 0, stream>>>(
      x, nullptr, W_attn, b_attn, nullptr, q_ws, k_ws, v_ws, MTOT, 3 * N_EMBD,
      N_EMBD);

  // 2) self_y = x @ W_self + b_self
  gemm_bf16_wmma<0, 0><<<dim3(MTOT / 64, N_EMBD / 128), blk, 0, stream>>>(
      x, nullptr, W_self, b_self, self_y, nullptr, nullptr, nullptr, MTOT,
      N_EMBD, N_EMBD);

  // 3) att_y = (sigmoid(QK^T/8) * coulomb) @ V
  attn_sigmoid_wmma<<<dim3(BZ * N_HEAD, TZ / 128), blk, 0, stream>>>(
      q_ws, k_ws, v_ws, coul, att_y);

  // 4) out = (att_y + self_y) @ W_proj + b_proj
  gemm_bf16_wmma<1, 0><<<dim3(MTOT / 64, N_EMBD / 128), blk, 0, stream>>>(
      att_y, self_y, W_proj, b_proj, out, nullptr, nullptr, nullptr, MTOT,
      N_EMBD, N_EMBD);
}
